// CausalSelfAttention_16063177687265
// MI455X (gfx1250) — matrix operand order repaired and checked by reading, not compiled, not run on hardware
//
#include <hip/hip_runtime.h>
#include <hip/hip_bf16.h>
#include <stdint.h>

typedef unsigned short u16;
typedef __bf16 bf16x16 __attribute__((ext_vector_type(16)));
typedef float  floatx8 __attribute__((ext_vector_type(8)));

union FragBF { bf16x16 f; uint4 q[2]; };
union V16U   { uint4 v[2]; u16 h[16]; };

__device__ __forceinline__ u16 f2bf(float x) {
  uint32_t u = __float_as_uint(x);
  u += 0x7FFFu + ((u >> 16) & 1u);   // round-to-nearest-even
  return (u16)(u >> 16);
}
__device__ __forceinline__ float bf2f(u16 h) {
  return __uint_as_float(((uint32_t)h) << 16);
}

// ---- DPP-based 16-lane reductions (pure VALU, no LDS/ds_bpermute) ----------
#define DPP8_SEL(a,b,c,d,e,f,g,h) \
  ((a)|((b)<<3)|((c)<<6)|((d)<<9)|((e)<<12)|((f)<<15)|((g)<<18)|((h)<<21))

template<int SEL>
__device__ __forceinline__ float dpp8f(float v) {
  return __int_as_float(__builtin_amdgcn_mov_dpp8(__float_as_int(v), SEL));
}
__device__ __forceinline__ float dpp_row_mirror(float v) {
  // DPP16 row_mirror (0x140): reverse the 16-lane row; after the dpp8 steps
  // each 8-group is uniform, so this combines the two 8-groups.
  return __int_as_float(
      __builtin_amdgcn_mov_dpp(__float_as_int(v), 0x140, 0xF, 0xF, true));
}
__device__ __forceinline__ float redmax16(float v) {
  v = fmaxf(v, dpp8f<DPP8_SEL(1,0,3,2,5,4,7,6)>(v));   // xor 1
  v = fmaxf(v, dpp8f<DPP8_SEL(2,3,0,1,6,7,4,5)>(v));   // xor 2
  v = fmaxf(v, dpp8f<DPP8_SEL(4,5,6,7,0,1,2,3)>(v));   // xor 4
  v = fmaxf(v, dpp_row_mirror(v));                     // cross 8-group
  return v;
}
__device__ __forceinline__ float redsum16(float v) {
  v += dpp8f<DPP8_SEL(1,0,3,2,5,4,7,6)>(v);
  v += dpp8f<DPP8_SEL(2,3,0,1,6,7,4,5)>(v);
  v += dpp8f<DPP8_SEL(4,5,6,7,0,1,2,3)>(v);
  v += dpp_row_mirror(v);
  return v;
}

// ---------------------------------------------------------------------------
// Tiled GEMM: C[M,N] = A[M,K] * B[K,N] + bias[N]
// BM=128, BN=128, BK=32; 256 threads = 8 waves; each wave: 32x64 (8 WMMA tiles)
// ---------------------------------------------------------------------------
template<int A_BF16, int OUT_BF16>
__global__ __launch_bounds__(256) void gemm_bias_wmma(
    const void* __restrict__ Ain, const float* __restrict__ Bw,
    const float* __restrict__ bias, void* __restrict__ Cout,
    int M, int N, int K)
{
  __shared__ __align__(16) u16 lA[128][32];
  __shared__ __align__(16) u16 lBT[128][32];
  const int tid  = threadIdx.x;
  const int lane = tid & 31;
  const int wid  = tid >> 5;
  const int wm   = wid >> 1;            // 0..3  -> 32 rows each
  const int wn   = wid & 1;             // 0..1  -> 64 cols each
  const int m0   = blockIdx.y * 128;
  const int n0   = blockIdx.x * 128;

  floatx8 acc[2][4];
#pragma unroll
  for (int a = 0; a < 2; ++a)
#pragma unroll
    for (int b = 0; b < 4; ++b)
      acc[a][b] = (floatx8){0,0,0,0,0,0,0,0};

  for (int k0 = 0; k0 < K; k0 += 32) {
    // ---- stage A tile (convert fp32 -> bf16 on the fly if needed)
    if (A_BF16) {
      const u16* A = (const u16*)Ain;
#pragma unroll
      for (int i = tid; i < 128 * 4; i += 256) {
        int r = i >> 2, c = (i & 3) * 8;
        *(uint4*)&lA[r][c] = *(const uint4*)(A + (size_t)(m0 + r) * K + k0 + c);
      }
    } else {
      const float* A = (const float*)Ain;
#pragma unroll
      for (int i = tid; i < 128 * 8; i += 256) {
        int r = i >> 3, c = (i & 7) * 4;
        float4 f = *(const float4*)(A + (size_t)(m0 + r) * K + k0 + c);
        u16* dst = &lA[r][c];
        dst[0] = f2bf(f.x); dst[1] = f2bf(f.y);
        dst[2] = f2bf(f.z); dst[3] = f2bf(f.w);
      }
    }
    // ---- stage B tile transposed: lBT[n][k]
#pragma unroll
    for (int i = tid; i < 32 * 32; i += 256) {
      int kk = i >> 5, n = (i & 31) * 4;
      float4 f = *(const float4*)(Bw + (size_t)(k0 + kk) * N + n0 + n);
      lBT[n + 0][kk] = f2bf(f.x); lBT[n + 1][kk] = f2bf(f.y);
      lBT[n + 2][kk] = f2bf(f.z); lBT[n + 3][kk] = f2bf(f.w);
    }
    __syncthreads();

    // ---- fragments + 8 WMMAs
    FragBF fa[2], fb[4];
    {
      const int r  = lane & 15;
      const int kh = (lane >> 4) * 8;
#pragma unroll
      for (int mt = 0; mt < 2; ++mt) {
        int rr = wm * 32 + mt * 16 + r;
        fa[mt].q[0] = *(const uint4*)&lA[rr][kh];
        fa[mt].q[1] = *(const uint4*)&lA[rr][16 + kh];
      }
      const int kb = (lane >> 4) * 8;
#pragma unroll
      for (int nt = 0; nt < 4; ++nt) {
        int nn = wn * 64 + nt * 16 + r;
        fb[nt].q[0] = *(const uint4*)&lBT[nn][kb];
        fb[nt].q[1] = *(const uint4*)&lBT[nn][kb + 16];
      }
    }
#pragma unroll
    for (int mt = 0; mt < 2; ++mt)
#pragma unroll
      for (int nt = 0; nt < 4; ++nt)
        acc[mt][nt] = __builtin_amdgcn_wmma_f32_16x16x32_bf16(
            false, fa[mt].f, false, fb[nt].f, (short)0, acc[mt][nt], false, false);
    __syncthreads();
  }

  // ---- epilogue
  const int hi = lane >> 4;
  const int cn = lane & 15;
#pragma unroll
  for (int mt = 0; mt < 2; ++mt) {
#pragma unroll
    for (int nt = 0; nt < 4; ++nt) {
      int gcol = n0 + wn * 64 + nt * 16 + cn;
      float bv = bias[gcol];
#pragma unroll
      for (int i = 0; i < 8; ++i) {
        int grow = m0 + wm * 32 + mt * 16 + i + 8 * hi;
        float v = acc[mt][nt][i] + bv;
        if (OUT_BF16) ((u16*)Cout)[(size_t)grow * N + gcol] = f2bf(v);
        else          ((float*)Cout)[(size_t)grow * N + gcol] = v;
      }
    }
  }
}

// ---------------------------------------------------------------------------
// RMSNorm + rotary + head-split: qkv bf16 [B,T,3,H,D] -> q,k,v bf16 [B,H,T,D]
// ---------------------------------------------------------------------------
__global__ __launch_bounds__(256) void rmsrope_split(
    const u16* __restrict__ qkv, const float* __restrict__ cs,
    const float* __restrict__ sn,
    u16* __restrict__ qo, u16* __restrict__ ko, u16* __restrict__ vo)
{
  const int lane = threadIdx.x & 31;
  const int gid  = blockIdx.x * 8 + (threadIdx.x >> 5);
  const int b    = gid >> 15;             // T*H = 32768
  const int rem  = gid & 32767;
  const int t    = rem >> 4;
  const int h    = rem & 15;
  const size_t base  = ((size_t)(b * 2048 + t) * 3) * 1024 + h * 64;
  const size_t obase = ((size_t)(b * 16 + h) * 2048 + t) * 64;
  const float c = cs[t * 32 + lane];
  const float s = sn[t * 32 + lane];

#pragma unroll
  for (int which = 0; which < 2; ++which) {   // 0=q, 1=k
    const u16* src = qkv + base + which * 1024;
    float x1 = bf2f(src[lane]);
    float x2 = bf2f(src[32 + lane]);
    float ss = x1 * x1 + x2 * x2;
    // reduce within the 16-lane rows with DPP, then combine the two rows
    ss = redsum16(ss);
    ss += __shfl_xor(ss, 16, 32);
    float r = rsqrtf(ss * (1.0f / 64.0f) + 1.1920929e-07f);
    x1 *= r; x2 *= r;
    float y1 =  x1 * c + x2 * s;
    float y2 = -x1 * s + x2 * c;
    u16* dst = (which == 0 ? qo : ko) + obase;
    dst[lane]      = f2bf(y1);
    dst[32 + lane] = f2bf(y2);
  }
  const u16* vs = qkv + base + 2048;
  vo[obase + lane]      = vs[lane];
  vo[obase + 32 + lane] = vs[32 + lane];
}

// ---------------------------------------------------------------------------
// Flash attention: block = 4 waves = 64 q-rows per (b,h); KV tiles of 32 keys.
// ---------------------------------------------------------------------------
__global__ __launch_bounds__(128) void flash_attn_wmma(
    const u16* __restrict__ Q, const u16* __restrict__ Kt,
    const u16* __restrict__ V, u16* __restrict__ Y)
{
  __shared__ __align__(16) u16 lVT[64][32];      // V^T tile [dim][key]
  __shared__ __align__(16) u16 lP[4][16][32];    // per-wave P tile
  const int tid  = threadIdx.x;
  const int lane = tid & 31;
  const int wid  = tid >> 5;
  const int qt   = blockIdx.x & 31;              // T/64 = 32
  const int bh   = blockIdx.x >> 5;              // B*H = 32
  const int b    = bh >> 4;
  const int h    = bh & 15;
  const int qb   = qt * 64;
  const int qrow0 = qb + wid * 16;

  const u16* Qh = Q  + (size_t)bh * 2048 * 64;
  const u16* Kh = Kt + (size_t)bh * 2048 * 64;
  const u16* Vh = V  + (size_t)bh * 2048 * 64;

  const int hi = lane >> 4;
  const int cn = lane & 15;

  // Q fragments (2 chunks of K=32), loaded once, straight from global
  FragBF qf[2];
  {
    int r  = qrow0 + cn;
    int kh = hi * 8;
#pragma unroll
    for (int cch = 0; cch < 2; ++cch) {
      qf[cch].q[0] = *(const uint4*)(Qh + (size_t)r * 64 + cch * 32 + kh);
      qf[cch].q[1] = *(const uint4*)(Qh + (size_t)r * 64 + cch * 32 + 16 + kh);
    }
  }

  floatx8 o[4];
#pragma unroll
  for (int nt = 0; nt < 4; ++nt) o[nt] = (floatx8){0,0,0,0,0,0,0,0};
  float m_[8], l_[8];
#pragma unroll
  for (int i = 0; i < 8; ++i) { m_[i] = -3.0e38f; l_[i] = 0.0f; }

  const int nkv = qb / 32 + 2;   // causal: only tiles with kb < qb+64

  for (int kvt = 0; kvt < nkv; ++kvt) {
    const int kb = kvt * 32;

    __syncthreads();  // previous V^T consumers done
    {
      // stage V^T: two b128 global loads per thread, scatter to LDS
      int key  = tid >> 2;
      int dseg = (tid & 3) * 16;
      V16U vv;
      const uint4* vsrc = (const uint4*)(Vh + (size_t)(kb + key) * 64 + dseg);
      vv.v[0] = vsrc[0];
      vv.v[1] = vsrc[1];
#pragma unroll
      for (int j = 0; j < 16; ++j) lVT[dseg + j][key] = vv.h[j];
    }
    __syncthreads();

    // ---- scores: 16x32 per wave (two 16x16 halves), K fragments from global
    floatx8 s[2];
#pragma unroll
    for (int half = 0; half < 2; ++half) {
      floatx8 sc = (floatx8){0,0,0,0,0,0,0,0};
      const u16* krow = Kh + (size_t)(kb + half * 16 + cn) * 64;
#pragma unroll
      for (int cch = 0; cch < 2; ++cch) {
        FragBF kf;
        int kd = cch * 32 + hi * 8;
        kf.q[0] = *(const uint4*)(krow + kd);
        kf.q[1] = *(const uint4*)(krow + kd + 16);
        sc = __builtin_amdgcn_wmma_f32_16x16x32_bf16(
            false, qf[cch].f, false, kf.f, (short)0, sc, false, false);
      }
      s[half] = sc;
    }

    // ---- causal mask + online softmax (rows live in 16-lane halves)
    float alpha[8];
#pragma unroll
    for (int i = 0; i < 8; ++i) {
      int row = qrow0 + i + 8 * hi;
      float s0 = s[0][i] * 0.125f;
      float s1 = s[1][i] * 0.125f;
      if (kb + cn > row)      s0 = -3.0e38f;
      if (kb + 16 + cn > row) s1 = -3.0e38f;
      float mc = redmax16(fmaxf(s0, s1));
      float mn = fmaxf(m_[i], mc);
      float a  = __expf(m_[i] - mn);
      float p0 = __expf(s0 - mn);
      float p1 = __expf(s1 - mn);
      float rs = redsum16(p0 + p1);
      l_[i] = l_[i] * a + rs;
      m_[i] = mn;
      alpha[i] = a;
      lP[wid][i + 8 * hi][cn]      = f2bf(p0);   // P tile [row][key]
      lP[wid][i + 8 * hi][16 + cn] = f2bf(p1);
    }
#pragma unroll
    for (int nt = 0; nt < 4; ++nt)
#pragma unroll
      for (int i = 0; i < 8; ++i)
        o[nt][i] *= alpha[i];

    // ---- PV: P as A-fragment from LDS, V^T as B-fragments
    FragBF pf;
    {
      int kh2 = hi * 8;
      pf.q[0] = *(const uint4*)&lP[wid][cn][kh2];
      pf.q[1] = *(const uint4*)&lP[wid][cn][16 + kh2];
    }
#pragma unroll
    for (int nt = 0; nt < 4; ++nt) {
      FragBF vf;
      int dcol = nt * 16 + cn;
      int kk   = hi * 8;
      vf.q[0] = *(const uint4*)&lVT[dcol][kk];
      vf.q[1] = *(const uint4*)&lVT[dcol][kk + 16];
      o[nt] = __builtin_amdgcn_wmma_f32_16x16x32_bf16(
          false, pf.f, false, vf.f, (short)0, o[nt], false, false);
    }
  }

  // ---- epilogue: y[b][row][h*64 + d] = O / l
#pragma unroll
  for (int i = 0; i < 8; ++i) {
    int row = qrow0 + i + 8 * hi;
    float inv = 1.0f / l_[i];
#pragma unroll
    for (int nt = 0; nt < 4; ++nt)
      Y[((size_t)b * 2048 + row) * 1024 + h * 64 + nt * 16 + cn] =
          f2bf(o[nt][i] * inv);
  }
}

// ---------------------------------------------------------------------------
extern "C" void kernel_launch(void* const* d_in, const int* in_sizes, int n_in,
                              void* d_out, int out_size, void* d_ws, size_t ws_size,
                              hipStream_t stream) {
  (void)in_sizes; (void)n_in; (void)out_size; (void)ws_size;
  const float* x      = (const float*)d_in[0];
  const float* cosp   = (const float*)d_in[1];
  const float* sinp   = (const float*)d_in[2];
  const float* W_attn = (const float*)d_in[3];
  const float* b_attn = (const float*)d_in[4];
  const float* W_proj = (const float*)d_in[5];
  const float* b_proj = (const float*)d_in[6];

  const int Bsz = 2, T = 2048, C = 1024, H = 16;
  const size_t MT = (size_t)Bsz * T;                 // 4096 rows

  // workspace layout (bf16), total ~56 MB
  u16* qkv = (u16*)d_ws;                             // [MT, 3C]
  u16* qb  = qkv + MT * 3 * C;                       // [B,H,T,D]
  u16* kb  = qb  + MT * C;
  u16* vb  = kb  + MT * C;
  u16* yb  = vb  + MT * C;                           // [B,T,C]

  gemm_bias_wmma<0, 1><<<dim3(3 * C / 128, MT / 128), 256, 0, stream>>>(
      x, W_attn, b_attn, qkv, (int)MT, 3 * C, C);

  rmsrope_split<<<dim3(Bsz * T * H / 8), 256, 0, stream>>>(
      qkv, cosp, sinp, qb, kb, vb);

  flash_attn_wmma<<<dim3(Bsz * H * (T / 64)), 128, 0, stream>>>(qb, kb, vb, yb);

  gemm_bias_wmma<1, 0><<<dim3(C / 128, MT / 128), 256, 0, stream>>>(
      yb, W_proj, b_proj, d_out, (int)MT, C, C);
}
